// VanillaRNN_3925600108561
// MI455X (gfx1250) — compile-verified, matrix operand order checked by reading, not run on hardware
//
#include <hip/hip_runtime.h>
#include <math.h>
#include <stdint.h>

#define BATCH 256
#define SEQ   512
#define HID   2048
#define NCLS  10

#define NBLOCKS  64
#define NTHREADS 256
#define HELEMS   (BATCH * HID)     // elements per h buffer
#define KS_ITERS (HID / 64)        // 32 k super-chunks (even)

typedef __attribute__((ext_vector_type(16))) __bf16 v16bf;
typedef __attribute__((ext_vector_type(8)))  float  v8f;

union BF16Tile { v16bf v; float4 q[2]; };

// ---------------------------------------------------------------------------
// CDNA5 async global->LDS copy (ASYNCcnt-tracked, no VGPR round-trip).
// LDS operand = low 32 bits of the flat address (aperture: addr[31:0]).
// ---------------------------------------------------------------------------
__device__ __forceinline__ void async_copy_b128(unsigned lds, const void* gaddr) {
    asm volatile("global_load_async_to_lds_b128 %0, %1, off"
                 :: "v"(lds), "v"(gaddr) : "memory");
}
__device__ __forceinline__ void wait_async0() {
#if __has_builtin(__builtin_amdgcn_s_wait_asynccnt)
    __builtin_amdgcn_s_wait_asynccnt(0);
#else
    asm volatile("s_wait_asynccnt 0x0" ::: "memory");
#endif
}
__device__ __forceinline__ unsigned lds_addr(const void* p) {
    return (unsigned)(uintptr_t)p;
}

// ---------------------------------------------------------------------------
// Device-wide split barrier (generation counter). All NBLOCKS blocks resident.
// ---------------------------------------------------------------------------
__device__ __forceinline__ void grid_barrier(int* cnt, int* gen) {
    __syncthreads();
    if (threadIdx.x == 0) {
        __threadfence();
        int g = __hip_atomic_load(gen, __ATOMIC_RELAXED, __HIP_MEMORY_SCOPE_AGENT);
        int c = __hip_atomic_fetch_add(cnt, 1, __ATOMIC_ACQ_REL, __HIP_MEMORY_SCOPE_AGENT);
        if (c == NBLOCKS - 1) {
            __hip_atomic_store(cnt, 0, __ATOMIC_RELAXED, __HIP_MEMORY_SCOPE_AGENT);
            __hip_atomic_fetch_add(gen, 1, __ATOMIC_RELEASE, __HIP_MEMORY_SCOPE_AGENT);
        } else {
            while (__hip_atomic_load(gen, __ATOMIC_ACQUIRE, __HIP_MEMORY_SCOPE_AGENT) == g) {
                __builtin_amdgcn_s_sleep(2);
            }
        }
        __threadfence();
    }
    __syncthreads();
}

// ---------------------------------------------------------------------------
// Setup: whh fp32 -> bf16 row-major; flag init.
// ---------------------------------------------------------------------------
__global__ __launch_bounds__(256) void cvt_whh_kernel(const float* __restrict__ whh,
                                                      void* whhb_v, int* flags) {
    __bf16* whhb = (__bf16*)whhb_v;
    int i = (blockIdx.x * 256 + threadIdx.x) * 4;
    float4 f = *(const float4*)(whh + i);
    union { __bf16 b[4]; uint2 u; } pk;
    pk.b[0] = (__bf16)f.x; pk.b[1] = (__bf16)f.y;
    pk.b[2] = (__bf16)f.z; pk.b[3] = (__bf16)f.w;
    *(uint2*)(whhb + i) = pk.u;
    if (blockIdx.x == 0 && threadIdx.x == 0) { flags[0] = 0; flags[1] = 0; }
}

// Setup: xT[t][b] = x[b][t]  (coalesced per-step column loads later)
__global__ __launch_bounds__(256) void xpose_kernel(const float* __restrict__ x,
                                                    float* __restrict__ xT) {
    int i = blockIdx.x * 256 + threadIdx.x;       // 131072
    int b = i >> 9, t = i & 511;
    xT[t * BATCH + b] = x[i];
}

// ---------------------------------------------------------------------------
// Persistent RNN: 512 timesteps + classifier in one launch.
// Block: 64 rows x 128 cols, 8 waves; wave: 32x32 output (2M x 2N tiles).
// Double-buffered LDS tiles filled by async global->LDS DMA; manual 2x
// unroll makes the ping-pong index compile-time constant.
// ---------------------------------------------------------------------------
__global__ __launch_bounds__(NTHREADS) void rnn_persist_kernel(
    const float* __restrict__ xT,  // [SEQ][BATCH]
    const float* __restrict__ whx, // [HID]
    const float* __restrict__ bh,  // [HID]
    const float* __restrict__ wph, // [NCLS][HID]
    const float* __restrict__ bp,  // [NCLS]
    float* __restrict__ out,       // [BATCH][NCLS]
    void* whhb_v, void* hb_v, int* flags) {

    const __bf16* __restrict__ whhb = (const __bf16*)whhb_v;
    __bf16* hb = (__bf16*)hb_v;                      // 2 x HELEMS ping-pong

    __shared__ __align__(16) __bf16 Ash[2][64 * 64];     // 16 KB
    __shared__ __align__(16) __bf16 Bsh[2][128 * 64];    // 32 KB
    __shared__ float Xsh[64];                            // x[:,t] slice

    const int tid  = threadIdx.x;
    const int lane = tid & 31;
    const int wave = tid >> 5;          // 0..7
    const int wr   = wave >> 2;         // 0..1 : 32-row group
    const int wc   = wave & 3;          // 0..3 : 32-col group

    const int rb = blockIdx.x >> 4;     // 0..3  -> 64-row slab
    const int cb = blockIdx.x & 15;     // 0..15 -> 128-col slab
    const int rowBase = rb * 64;
    const int colBase = cb * 128;

    const int mIdx = lane & 15;         // within-tile M/N index
    const int kh   = lane >> 4;         // K-half selector

    // staging coordinates: all chunks share c; rows step by 32
    const int sr = tid >> 3;            // 0..31
    const int sc = (tid & 7) * 8;       // 0,8,..,56

    // precomputed LDS destinations (both buffers, constants per thread)
    unsigned lA[2][2], lB[2][4];
    #pragma unroll
    for (int nb = 0; nb < 2; ++nb) {
        #pragma unroll
        for (int i = 0; i < 2; ++i) lA[nb][i] = lds_addr(&Ash[nb][(sr + 32 * i) * 64 + sc]);
        #pragma unroll
        for (int i = 0; i < 4; ++i) lB[nb][i] = lds_addr(&Bsh[nb][(sr + 32 * i) * 64 + sc]);
    }
    // step-invariant B source base pointers
    const __bf16* pB[4];
    #pragma unroll
    for (int i = 0; i < 4; ++i) pB[i] = whhb + (colBase + sr + 32 * i) * HID + sc;

    // ---- zero-init h ping buffer (read at t=0) ----
    {
        unsigned gt = blockIdx.x * NTHREADS + tid;     // 16384 threads
        uint4 z = {0u, 0u, 0u, 0u};
        uint4* p = (uint4*)hb_v;                        // 65536 chunks
        #pragma unroll
        for (int i = 0; i < 4; ++i) p[gt * 4 + i] = z;
    }

    // ---- per-lane loop-invariant epilogue constants ----
    const int col0 = colBase + wc * 32 + mIdx;
    const int col1 = col0 + 16;
    const float bh0 = bh[col0], bh1 = bh[col1];
    const float wx0 = whx[col0], wx1 = whx[col1];
    const int mbase0 = rowBase + wr * 32 + kh * 8;   // rows of A0 tile in C/D
    const int mbase1 = mbase0 + 16;                  // rows of A1 tile

    int cur = 0;
    grid_barrier(&flags[0], &flags[1]);

    for (int t = 0; t < SEQ; ++t) {
        const __bf16* hr = hb + cur * HELEMS;
        __bf16*       hw = hb + (cur ^ 1) * HELEMS;

        if (tid < 64) Xsh[tid] = xT[t * BATCH + rowBase + tid];

        v8f acc00 = {0.f,0.f,0.f,0.f,0.f,0.f,0.f,0.f};
        v8f acc01 = acc00, acc10 = acc00, acc11 = acc00;

        // running source pointers (advance by 64 elems per issued chunk)
        const __bf16* qa0 = hr + (rowBase + sr) * HID + sc;
        const __bf16* qa1 = qa0 + 32 * HID;
        const __bf16* qb0 = pB[0];
        const __bf16* qb1 = pB[1];
        const __bf16* qb2 = pB[2];
        const __bf16* qb3 = pB[3];

        // issue one 64-k chunk into buffer nb from current pointers
        auto issue = [&](int nb) {
            async_copy_b128(lA[nb][0], qa0);
            async_copy_b128(lA[nb][1], qa1);
            async_copy_b128(lB[nb][0], qb0);
            async_copy_b128(lB[nb][1], qb1);
            async_copy_b128(lB[nb][2], qb2);
            async_copy_b128(lB[nb][3], qb3);
            qa0 += 64; qa1 += 64;
            qb0 += 64; qb1 += 64; qb2 += 64; qb3 += 64;
        };

        auto compute = [&](int buf) {
            #pragma unroll
            for (int kc = 0; kc < 64; kc += 32) {
                BF16Tile a0, a1, b0, b1;
                const float4* ap0 = (const float4*)(&Ash[buf][(wr * 32 + mIdx) * 64 + kc + kh * 16]);
                const float4* ap1 = (const float4*)(&Ash[buf][(wr * 32 + 16 + mIdx) * 64 + kc + kh * 16]);
                a0.q[0] = ap0[0]; a0.q[1] = ap0[1];
                a1.q[0] = ap1[0]; a1.q[1] = ap1[1];
                const float4* bq0 = (const float4*)(&Bsh[buf][(wc * 32 + mIdx) * 64 + kc + kh * 16]);
                const float4* bq1 = (const float4*)(&Bsh[buf][(wc * 32 + 16 + mIdx) * 64 + kc + kh * 16]);
                b0.q[0] = bq0[0]; b0.q[1] = bq0[1];
                b1.q[0] = bq1[0]; b1.q[1] = bq1[1];
                acc00 = __builtin_amdgcn_wmma_f32_16x16x32_bf16(
                            false, a0.v, false, b0.v, (short)0, acc00, false, false);
                acc01 = __builtin_amdgcn_wmma_f32_16x16x32_bf16(
                            false, a0.v, false, b1.v, (short)0, acc01, false, false);
                acc10 = __builtin_amdgcn_wmma_f32_16x16x32_bf16(
                            false, a1.v, false, b0.v, (short)0, acc10, false, false);
                acc11 = __builtin_amdgcn_wmma_f32_16x16x32_bf16(
                            false, a1.v, false, b1.v, (short)0, acc11, false, false);
            }
        };

        issue(0);                         // prologue: chunk 0 -> buf0

        for (int it = 0; it < KS_ITERS; it += 2) {
            // even half: compute buf0, DMA chunk it+1 -> buf1
            wait_async0();
            __syncthreads();
            if (it + 1 < KS_ITERS) issue(1);
            compute(0);
            // odd half: compute buf1, DMA chunk it+2 -> buf0
            wait_async0();
            __syncthreads();
            if (it + 2 < KS_ITERS) issue(0);
            compute(1);
        }

        // ---- epilogue: h_new = tanh(acc + bh + x_t * whx), store bf16 ----
        #pragma unroll
        for (int v = 0; v < 8; ++v) {
            int m0 = mbase0 + v, m1 = mbase1 + v;
            float xv0 = Xsh[m0 - rowBase];
            float xv1 = Xsh[m1 - rowBase];
            hw[m0 * HID + col0] = (__bf16)tanhf(acc00[v] + bh0 + xv0 * wx0);
            hw[m0 * HID + col1] = (__bf16)tanhf(acc01[v] + bh1 + xv0 * wx1);
            hw[m1 * HID + col0] = (__bf16)tanhf(acc10[v] + bh0 + xv1 * wx0);
            hw[m1 * HID + col1] = (__bf16)tanhf(acc11[v] + bh1 + xv1 * wx1);
        }

        cur ^= 1;
        grid_barrier(&flags[0], &flags[1]);
    }

    // ---- classifier: p = h_last @ wph^T + bp (fp32 accumulation) ----
    if (blockIdx.x < NCLS && tid < BATCH) {
        const __bf16* hl = hb + cur * HELEMS;   // cur == 0 after 512 flips
        int c = blockIdx.x, b = tid;
        float s = bp[c];
        const __bf16* hrow = hl + b * HID;
        const float*  wrow = wph + c * HID;
        for (int j = 0; j < HID; ++j) s += (float)hrow[j] * wrow[j];
        out[b * NCLS + c] = s;
    }
}

// ---------------------------------------------------------------------------
// Host launcher
// ---------------------------------------------------------------------------
extern "C" void kernel_launch(void* const* d_in, const int* in_sizes, int n_in,
                              void* d_out, int out_size, void* d_ws, size_t ws_size,
                              hipStream_t stream) {
    const float* x   = (const float*)d_in[0];  // [256][512]
    const float* whx = (const float*)d_in[1];  // [2048]
    const float* whh = (const float*)d_in[2];  // [2048][2048]
    const float* bh  = (const float*)d_in[3];  // [2048]
    const float* wph = (const float*)d_in[4];  // [10][2048]
    const float* bp  = (const float*)d_in[5];  // [10]

    char* ws = (char*)d_ws;
    void*  whhb = (void*)(ws);                              // 8 MiB bf16 whh
    void*  hbuf = (void*)(ws + (size_t)(8u << 20));         // 2 MiB h ping+pong
    int*   flg  = (int*)(ws + (size_t)(10u << 20));         // barrier cnt/gen
    float* xT   = (float*)(ws + (size_t)(10u << 20) + 4096); // 512 KiB x^T

    cvt_whh_kernel<<<4096, 256, 0, stream>>>(whh, whhb, flg);
    xpose_kernel<<<512, 256, 0, stream>>>(x, xT);
    rnn_persist_kernel<<<NBLOCKS, NTHREADS, 0, stream>>>(
        xT, whx, bh, wph, bp, (float*)d_out, whhb, hbuf, flg);
}
